// InvariantPointAttention_7988639171223
// MI455X (gfx1250) — matrix operand order repaired and checked by reading, not compiled, not run on hardware
//
#include <hip/hip_runtime.h>
#include <hip/hip_bf16.h>
#include <math.h>
#include <stdint.h>

// ---------------------------------------------------------------------------
// Invariant Point Attention forward for MI455X (gfx1250), wave32 + WMMA bf16
// + async global->LDS double-buffered staging in the attention inner loop.
// ---------------------------------------------------------------------------

typedef __attribute__((ext_vector_type(16))) __bf16 v16bf;
typedef __attribute__((ext_vector_type(8)))  __bf16 v8bf;
typedef __attribute__((ext_vector_type(8)))  float  v8f;

constexpr int kB   = 4;
constexpr int kN   = 1024;
constexpr int kDIM = 512;
constexpr int kH   = 8;
constexpr int kDH  = 64;
constexpr int kP   = 4;
constexpr int kBN  = kB * kN;                  // 4096 token rows
constexpr int kQK  = 96;                       // augmented q/k feature dim
constexpr int kVD  = 80;                       // augmented v dim
constexpr int kFIN = kH * (kDH + kP * 4);      // 640 concat features
constexpr float kSCALAR_SCALE = 0.08838834764831845f;   // (2*64)^-0.5
constexpr float kPOINT_SCALE  = 0.16666666666666666f;   // (2*4*4.5)^-0.5
constexpr float kEPS = 1e-8f;

// ---------------- workspace layout (bytes) ----------------
constexpr size_t SZ_XB    = (size_t)kBN * kDIM * 2;
constexpr size_t SZ_WSC   = (size_t)kDIM * 512 * 2;
constexpr size_t SZ_WPT   = (size_t)kDIM * 96 * 2;
constexpr size_t SZ_WO    = (size_t)kFIN * kDIM * 2;
constexpr size_t SZ_QAUG  = (size_t)kB * kH * kN * kQK * 2;   // 6 MB
constexpr size_t SZ_VAUG  = (size_t)kB * kH * kN * kVD * 2;   // 5 MB
constexpr size_t SZ_SQ    = (size_t)kB * kH * kN * 4;
constexpr size_t SZ_TP    = (size_t)kBN * 96 * 4;
constexpr size_t SZ_KFRAG = SZ_QAUG;
constexpr size_t SZ_VFRAG = SZ_VAUG;
constexpr size_t SZ_RES   = (size_t)kB * kH * kN * kVD * 4;   // 10.5 MB
constexpr size_t SZ_AFIN  = (size_t)kBN * kFIN * 2;           // 5 MB

constexpr size_t OFF_XB    = 0;
constexpr size_t OFF_WQS   = OFF_XB   + SZ_XB;
constexpr size_t OFF_WKS   = OFF_WQS  + SZ_WSC;
constexpr size_t OFF_WVS   = OFF_WKS  + SZ_WSC;
constexpr size_t OFF_WQP   = OFF_WVS  + SZ_WSC;
constexpr size_t OFF_WKP   = OFF_WQP  + SZ_WPT;
constexpr size_t OFF_WVP   = OFF_WKP  + SZ_WPT;
constexpr size_t OFF_WO    = OFF_WVP  + SZ_WPT;
constexpr size_t OFF_QAUG  = OFF_WO   + SZ_WO;
constexpr size_t OFF_KAUG  = OFF_QAUG + SZ_QAUG;
constexpr size_t OFF_VAUG  = OFF_KAUG + SZ_QAUG;
constexpr size_t OFF_SQK   = OFF_VAUG + SZ_VAUG;
constexpr size_t OFF_SQQ   = OFF_SQK  + SZ_SQ;
constexpr size_t OFF_TMP   = OFF_SQQ  + SZ_SQ;
// phase 1 temps (dead after k_assemble_points):
constexpr size_t OFF_TQP   = OFF_TMP;
constexpr size_t OFF_TKP   = OFF_TQP + SZ_TP;
constexpr size_t OFF_TVP   = OFF_TKP + SZ_TP;
// phase 2 (sequential stream order makes the aliasing safe):
constexpr size_t OFF_KFRAG = OFF_TMP;                 // over dead TQP/TKP/TVP
constexpr size_t OFF_VFRAG = OFF_KFRAG + SZ_KFRAG;
constexpr size_t OFF_AFIN  = OFF_VFRAG + SZ_VFRAG;
constexpr size_t OFF_RES   = OFF_KAUG;                // over dead KAUG/VAUG

// ---------------- WMMA fragment helpers (bf16 16x16x32 layouts as used by this code) ----
// A 16x32 (16-bit): lane L: row M = L%16 ; element e -> K = 16*(L/16) + e
// B 32x16 (16-bit): lane L: col N = L%16 ; element e -> K = 16*(L/16) + e
// C/D 16x16 (f32):  lane L: col N = L%16 ; vgpr r   -> row M = r + 8*(L/16)

__device__ __forceinline__ v16bf loadAfrag(const __bf16* rowptr, int k0, int hf) {
  const v8bf lo = *reinterpret_cast<const v8bf*>(rowptr + k0 + 16 * hf);
  const v8bf hi = *reinterpret_cast<const v8bf*>(rowptr + k0 + 16 * hf + 8);
  v16bf out;
#pragma unroll
  for (int e = 0; e < 8; ++e) { out[e] = lo[e]; out[e + 8] = hi[e]; }
  return out;
}

// Fragment-ordered B tile: lane L's 16 elements are contiguous at p + L*16.
__device__ __forceinline__ v16bf loadBfragPacked(const __bf16* tilebase, int lane) {
  return *reinterpret_cast<const v16bf*>(tilebase + (size_t)lane * 16);
}

__device__ __forceinline__ v8f wmma_bf16(v16bf a, v16bf b, v8f c) {
  return __builtin_amdgcn_wmma_f32_16x16x32_bf16(false, a, false, b,
                                                 (short)0, c, false, false);
}

// ---------------- kernels ----------------

__global__ void k_f32_to_bf16(const float* __restrict__ in, __bf16* __restrict__ out, int n) {
  int i = blockIdx.x * blockDim.x + threadIdx.x;
  if (i < n) out[i] = (__bf16)in[i];
}

// Convert a K x N fp32 weight matrix into bf16 WMMA-B-fragment order:
// Wf[((k0/32)*(N/16) + nt)*512 + L*16 + e] = W[(k0 + 16*(L/16) + e)*N + nt*16 + L%16]
__global__ void k_pack_w(const float* __restrict__ W, __bf16* __restrict__ Wf,
                         int Ncols, int total) {
  int idx = blockIdx.x * blockDim.x + threadIdx.x;
  if (idx >= total) return;
  const int e = idx & 15;
  const int L = (idx >> 4) & 31;
  const int rest = idx >> 9;
  const int ntn = Ncols >> 4;
  const int nt = rest % ntn;
  const int kc = rest / ntn;
  const int k = kc * 32 + 16 * (L >> 4) + e;
  const int n = nt * 16 + (L & 15);
  Wf[idx] = (__bf16)W[(size_t)k * Ncols + n];
}

// Repack kaug rows ([bh][n][96]) into per-(bh, 32-row chunk) fragment slabs:
// kfrag[((bh*32+jc)*6 + jt*3 + t)*512 + L*16 + e]
//   = kaug[bh][jc*32 + jt*16 + L%16][32*t + 16*(L/16) + e]
__global__ void k_pack_kfrag(const __bf16* __restrict__ kaug, __bf16* __restrict__ kfrag) {
  int idx = blockIdx.x * blockDim.x + threadIdx.x;
  if (idx >= kB * kH * kN * kQK) return;
  const int e = idx & 15;
  const int L = (idx >> 4) & 31;
  const int s = (idx >> 9) % 6;
  const int cj = (idx >> 9) / 6;
  const int bh = cj >> 5, jc = cj & 31;
  const int jt = s / 3, t = s % 3;
  const int n = jc * 32 + jt * 16 + (L & 15);
  const int f = 32 * t + 16 * (L >> 4) + e;
  kfrag[idx] = kaug[((size_t)bh * kN + n) * kQK + f];
}

// vfrag[((bh*32+jc)*5 + t)*512 + L*16 + e]
//   = vaug[bh][jc*32 + 16*(L/16) + e][16*t + L%16]
__global__ void k_pack_vfrag(const __bf16* __restrict__ vaug, __bf16* __restrict__ vfrag) {
  int idx = blockIdx.x * blockDim.x + threadIdx.x;
  if (idx >= kB * kH * kN * kVD) return;
  const int e = idx & 15;
  const int L = (idx >> 4) & 31;
  const int t = (idx >> 9) % 5;
  const int cj = (idx >> 9) / 5;
  const int bh = cj >> 5, jc = cj & 31;
  const int n = jc * 32 + 16 * (L >> 4) + e;
  const int d = 16 * t + (L & 15);
  vfrag[idx] = vaug[((size_t)bh * kN + n) * kVD + d];
}

// Generic bf16 WMMA GEMM with fragment-packed W. 4 waves/block; each wave
// computes a 16 x (TN*16) tile, reusing one A-fragment across TN WMMAs.
// mode 0: fp32 row-major C (+ optional bias[col]).
// mode 1: scatter bf16 scaled output into augmented [bh][n][augstride] tensor.
template <int TN>
__global__ __launch_bounds__(128) void k_gemm(
    const __bf16* __restrict__ A, const __bf16* __restrict__ Wf,
    const float* __restrict__ bias, float* __restrict__ Cf,
    __bf16* __restrict__ Caug, int mode, float scale, int augstride,
    int M, int Ncols, int K) {
  const int lane = threadIdx.x & 31, wid = threadIdx.x >> 5;
  const int tile = blockIdx.x * 4 + wid;
  const int ntn  = Ncols >> 4;          // 16-col tiles
  const int ngrp = ntn / TN;            // col groups per row
  const int row0 = (tile / ngrp) << 4;
  const int ct0  = (tile % ngrp) * TN;  // first 16-col tile of this group
  const int hf = lane >> 4, l16 = lane & 15;
  const __bf16* arow = A + (size_t)(row0 + l16) * K;
  const size_t wstride = (size_t)ntn * 512;      // one 32-wide k chunk
  const __bf16* wbase = Wf + (size_t)ct0 * 512;
  v8f acc[TN];
  v8f zero = {};
#pragma unroll
  for (int j = 0; j < TN; ++j) acc[j] = zero;
  for (int k0 = 0; k0 < K; k0 += 32) {
    __builtin_prefetch(arow + k0 + 256, 0, 3);   // global_prefetch_b8, WGP scope
    v16bf a = loadAfrag(arow, k0, hf);
    const __bf16* wk = wbase + (size_t)(k0 >> 5) * wstride;
#pragma unroll
    for (int j = 0; j < TN; ++j) {
      v16bf b = loadBfragPacked(wk + (size_t)j * 512, lane);
      acc[j] = wmma_bf16(a, b, acc[j]);
    }
  }
#pragma unroll
  for (int j = 0; j < TN; ++j) {
    const int col = (ct0 + j) * 16 + l16;
    if (mode == 0) {
      const float bb = bias ? bias[col] : 0.0f;
#pragma unroll
      for (int r = 0; r < 8; ++r) {
        int row = row0 + r + 8 * hf;
        Cf[(size_t)row * Ncols + col] = acc[j][r] + bb;
      }
    } else {
      const int h = col >> 6, d = col & 63;
#pragma unroll
      for (int r = 0; r < 8; ++r) {
        int row = row0 + r + 8 * hf;
        int bb_ = row >> 10, n = row & (kN - 1);
        Caug[((size_t)(bb_ * kH + h) * kN + n) * augstride + d] =
            (__bf16)(scale * acc[j][r]);
      }
    }
  }
}

// Rotate/translate the point projections into the global frame and build the
// point slices of the augmented q/k/v tensors; emit squared norms for biases.
__global__ void k_assemble_points(
    const float* __restrict__ tqp, const float* __restrict__ tkp,
    const float* __restrict__ tvp, const float* __restrict__ rot,
    const float* __restrict__ trans, const float* __restrict__ pweights,
    __bf16* __restrict__ qaug, __bf16* __restrict__ kaug, __bf16* __restrict__ vaug,
    float* __restrict__ sqk, float* __restrict__ sqq) {
  int idx = blockIdx.x * blockDim.x + threadIdx.x;   // (b*N+n)*H + h
  if (idx >= kBN * kH) return;
  const int h  = idx & (kH - 1);
  const int bn = idx >> 3;
  const int b  = bn >> 10, n = bn & (kN - 1);
  const float pw = log1pf(__expf(pweights[h]));      // softplus
  const float cq = pw * kPOINT_SCALE;
  const float* R = rot + (size_t)bn * 9;
  const float* t = trans + (size_t)bn * 3;
  const size_t arow = (size_t)(b * kH + h) * kN + n;
  __bf16* q = qaug + arow * kQK;
  __bf16* k = kaug + arow * kQK;
  __bf16* v = vaug + arow * kVD;
  const float* qp = tqp + (size_t)bn * 96 + h * 12;
  const float* kp = tkp + (size_t)bn * 96 + h * 12;
  const float* vp = tvp + (size_t)bn * 96 + h * 12;
  float ssq = 0.0f, ssk = 0.0f;
#pragma unroll
  for (int p = 0; p < kP; ++p) {
#pragma unroll
    for (int c = 0; c < 3; ++c) {
      const float* Rc = R + c * 3;
      float qg = Rc[0] * qp[p*3+0] + Rc[1] * qp[p*3+1] + Rc[2] * qp[p*3+2] + t[c];
      float kg = Rc[0] * kp[p*3+0] + Rc[1] * kp[p*3+1] + Rc[2] * kp[p*3+2] + t[c];
      float vg = Rc[0] * vp[p*3+0] + Rc[1] * vp[p*3+1] + Rc[2] * vp[p*3+2] + t[c];
      q[kDH + p * 3 + c] = (__bf16)(cq * qg);
      k[kDH + p * 3 + c] = (__bf16)kg;
      v[kDH + p * 3 + c] = (__bf16)vg;
      ssq += qg * qg;
      ssk += kg * kg;
    }
  }
#pragma unroll
  for (int f = kDH + 12; f < kQK; ++f) { q[f] = (__bf16)0.0f; k[f] = (__bf16)0.0f; }
#pragma unroll
  for (int f = kDH + 12; f < kVD; ++f) { v[f] = (__bf16)0.0f; }
  sqq[arow] = ssq;
  sqk[arow] = ssk;
}

// Issue async global->LDS copies of one (bh, j-chunk) K slab (6144 B) and
// V slab (5120 B). Contiguous in the packed layout -> b128 chunks.
__device__ __forceinline__ void async_stage(const __bf16* kslab, const __bf16* vslab,
                                            __bf16* kb, __bf16* vb, int tid) {
  const uint32_t koff = (uint32_t)(uintptr_t)kb;
  const uint32_t voff = (uint32_t)(uintptr_t)vb;
  const char* kg = (const char*)kslab;
  const char* vg = (const char*)vslab;
#pragma unroll
  for (int c = 0; c < 3; ++c) {
    const uint32_t off = (uint32_t)(tid + c * 128) * 16u;   // 384 x 16B = 6144B
    asm volatile("global_load_async_to_lds_b128 %0, %1, off"
                 :: "v"(koff + off), "v"(kg + off) : "memory");
  }
#pragma unroll
  for (int c = 0; c < 3; ++c) {
    const uint32_t off = (uint32_t)(tid + c * 128) * 16u;   // 320 x 16B = 5120B
    if (off < 5120u)
      asm volatile("global_load_async_to_lds_b128 %0, %1, off"
                   :: "v"(voff + off), "v"(vg + off) : "memory");
  }
}

// Flash-style attention: 4 waves/block share one (b,h); each wave owns a
// 16-row query tile. K/V fragment slabs are async-staged to LDS with double
// buffering; QK^T = 3 WMMAs, online softmax via 16-lane shuffles, P goes
// through LDS into A-fragment form, attn*V = 5 WMMAs.
__global__ __launch_bounds__(128) void k_attention(
    const __bf16* __restrict__ qaug, const __bf16* __restrict__ kfrag,
    const __bf16* __restrict__ vfrag, const float* __restrict__ sqk,
    const float* __restrict__ sqq, const float* __restrict__ pweights,
    float* __restrict__ res) {
  __shared__ __align__(32) __bf16 Kb[2][6 * 512];
  __shared__ __align__(32) __bf16 Vb[2][5 * 512];
  __shared__ __align__(32) __bf16 Pbuf[4][16][32];
  const int tid = threadIdx.x;
  const int lane = tid & 31, wid = tid >> 5;
  const int bh = blockIdx.x >> 4;                    // uniform per block
  const int it = (blockIdx.x & 15) * 4 + wid;
  const int h  = bh & (kH - 1);
  const int i0 = it << 4;
  const int hf = lane >> 4, l16 = lane & 15;
  const float pw  = log1pf(__expf(pweights[h]));
  const float cpt = -0.5f * pw * kPOINT_SCALE;
  const __bf16* qbase  = qaug + ((size_t)bh * kN + i0) * kQK;
  const __bf16* kfbase = kfrag + (size_t)bh * (32 * 6 * 512);
  const __bf16* vfbase = vfrag + (size_t)bh * (32 * 5 * 512);
  const float*  sqkb   = sqk + (size_t)bh * kN;
  const float*  sqqb   = sqq + (size_t)bh * kN;

  float rowb[8];
#pragma unroll
  for (int r = 0; r < 8; ++r) rowb[r] = cpt * sqqb[i0 + r + 8 * hf];

  v16bf qf[3];
  const __bf16* qrow = qbase + (size_t)l16 * kQK;
#pragma unroll
  for (int t = 0; t < 3; ++t) qf[t] = loadAfrag(qrow, 32 * t, hf);

  v8f zero = {};
  v8f acc[5];
#pragma unroll
  for (int t = 0; t < 5; ++t) acc[t] = zero;
  float rmax[8], rsum[8];
#pragma unroll
  for (int r = 0; r < 8; ++r) { rmax[r] = -3.0e38f; rsum[r] = 0.0f; }

  async_stage(kfbase, vfbase, &Kb[0][0], &Vb[0][0], tid);

  for (int jc = 0; jc < 32; ++jc) {
    const int cur = jc & 1;
    const int j0 = jc * 32;
    asm volatile("s_wait_asynccnt 0x0" ::: "memory");
    __syncthreads();                                 // staged slab visible
    if (jc + 1 < 32)
      async_stage(kfbase + (size_t)(jc + 1) * (6 * 512),
                  vfbase + (size_t)(jc + 1) * (5 * 512),
                  &Kb[1 - cur][0], &Vb[1 - cur][0], tid);

    v8f s0 = zero, s1 = zero;
#pragma unroll
    for (int t = 0; t < 3; ++t) {
      v16bf b0 = loadBfragPacked(&Kb[cur][(0 * 3 + t) * 512], lane);
      v16bf b1 = loadBfragPacked(&Kb[cur][(1 * 3 + t) * 512], lane);
      s0 = wmma_bf16(qf[t], b0, s0);
      s1 = wmma_bf16(qf[t], b1, s1);
    }
    const float cb0 = cpt * sqkb[j0 + l16];
    const float cb1 = cpt * sqkb[j0 + 16 + l16];
    float alpha[8];
#pragma unroll
    for (int r = 0; r < 8; ++r) {
      float v0 = s0[r] + cb0 + rowb[r];
      float v1 = s1[r] + cb1 + rowb[r];
      float m = fmaxf(v0, v1);
#pragma unroll
      for (int msk = 1; msk < 16; msk <<= 1) m = fmaxf(m, __shfl_xor(m, msk, 32));
      float nm = fmaxf(rmax[r], m);
      alpha[r] = __expf(rmax[r] - nm);
      rmax[r] = nm;
      float p0 = __expf(v0 - nm), p1 = __expf(v1 - nm);
      float ps = p0 + p1;
#pragma unroll
      for (int msk = 1; msk < 16; msk <<= 1) ps += __shfl_xor(ps, msk, 32);
      rsum[r] = rsum[r] * alpha[r] + ps;
      const int row = r + 8 * hf;
      Pbuf[wid][row][l16]      = (__bf16)p0;
      Pbuf[wid][row][16 + l16] = (__bf16)p1;
    }
    __syncthreads();
    v16bf pa = loadAfrag(&Pbuf[wid][l16][0], 0, hf);
#pragma unroll
    for (int t = 0; t < 5; ++t) {
#pragma unroll
      for (int r = 0; r < 8; ++r) acc[t][r] *= alpha[r];
      v16bf vb = loadBfragPacked(&Vb[cur][t * 512], lane);
      acc[t] = wmma_bf16(pa, vb, acc[t]);
    }
    __syncthreads();
  }

  float* rbase = res + ((size_t)bh * kN + i0) * kVD;
#pragma unroll
  for (int t = 0; t < 5; ++t) {
#pragma unroll
    for (int r = 0; r < 8; ++r) {
      const int row = r + 8 * hf;
      rbase[(size_t)row * kVD + 16 * t + l16] = acc[t][r] / rsum[r];
    }
  }
}

// Inverse-rotate the point outputs, compute norms, build the bf16 concat matrix.
__global__ void k_finalize(const float* __restrict__ res,
                           const float* __restrict__ rot,
                           const float* __restrict__ trans,
                           __bf16* __restrict__ afin) {
  int idx = blockIdx.x * blockDim.x + threadIdx.x;   // (b*N+n)*H + h
  if (idx >= kBN * kH) return;
  const int h  = idx & (kH - 1);
  const int bn = idx >> 3;
  const int b  = bn >> 10, n = bn & (kN - 1);
  const float* r = res + ((size_t)(b * kH + h) * kN + n) * kVD;
  __bf16* arow = afin + (size_t)bn * kFIN;
#pragma unroll 8
  for (int d = 0; d < kDH; ++d) arow[h * kDH + d] = (__bf16)r[d];
  const float* R = rot + (size_t)bn * 9;
  const float* t = trans + (size_t)bn * 3;
#pragma unroll
  for (int p = 0; p < kP; ++p) {
    const float g0 = r[kDH + p * 3 + 0] - t[0];
    const float g1 = r[kDH + p * 3 + 1] - t[1];
    const float g2 = r[kDH + p * 3 + 2] - t[2];
    const float l0 = R[0] * g0 + R[3] * g1 + R[6] * g2;   // R^T * g
    const float l1 = R[1] * g0 + R[4] * g1 + R[7] * g2;
    const float l2 = R[2] * g0 + R[5] * g1 + R[8] * g2;
    arow[kH * kDH + h * 12 + p * 3 + 0] = (__bf16)l0;
    arow[kH * kDH + h * 12 + p * 3 + 1] = (__bf16)l1;
    arow[kH * kDH + h * 12 + p * 3 + 2] = (__bf16)l2;
    arow[kH * kDH + 96 + h * 4 + p] =
        (__bf16)sqrtf(l0 * l0 + l1 * l1 + l2 * l2 + kEPS);
  }
}

// ---------------- launcher ----------------

extern "C" void kernel_launch(void* const* d_in, const int* in_sizes, int n_in,
                              void* d_out, int out_size, void* d_ws, size_t ws_size,
                              hipStream_t stream) {
  (void)in_sizes; (void)n_in; (void)out_size; (void)ws_size;
  const float* x     = (const float*)d_in[0];
  const float* rot   = (const float*)d_in[1];
  const float* trans = (const float*)d_in[2];
  const float* w_qs  = (const float*)d_in[3];
  const float* w_ks  = (const float*)d_in[4];
  const float* w_vs  = (const float*)d_in[5];
  const float* w_qp  = (const float*)d_in[6];
  const float* w_kp  = (const float*)d_in[7];
  const float* w_vp  = (const float*)d_in[8];
  const float* pwts  = (const float*)d_in[9];
  const float* w_o   = (const float*)d_in[10];
  const float* b_o   = (const float*)d_in[11];
  float* out = (float*)d_out;

  char* ws = (char*)d_ws;
  __bf16* xb    = (__bf16*)(ws + OFF_XB);
  __bf16* wqsf  = (__bf16*)(ws + OFF_WQS);
  __bf16* wksf  = (__bf16*)(ws + OFF_WKS);
  __bf16* wvsf  = (__bf16*)(ws + OFF_WVS);
  __bf16* wqpf  = (__bf16*)(ws + OFF_WQP);
  __bf16* wkpf  = (__bf16*)(ws + OFF_WKP);
  __bf16* wvpf  = (__bf16*)(ws + OFF_WVP);
  __bf16* wof   = (__bf16*)(ws + OFF_WO);
  __bf16* qaug  = (__bf16*)(ws + OFF_QAUG);
  __bf16* kaug  = (__bf16*)(ws + OFF_KAUG);
  __bf16* vaug  = (__bf16*)(ws + OFF_VAUG);
  float*  sqk   = (float*)(ws + OFF_SQK);
  float*  sqq   = (float*)(ws + OFF_SQQ);
  float*  tqp   = (float*)(ws + OFF_TQP);
  float*  tkp   = (float*)(ws + OFF_TKP);
  float*  tvp   = (float*)(ws + OFF_TVP);
  __bf16* kfrag = (__bf16*)(ws + OFF_KFRAG);
  __bf16* vfrag = (__bf16*)(ws + OFF_VFRAG);
  __bf16* afin  = (__bf16*)(ws + OFF_AFIN);
  float*  resb  = (float*)(ws + OFF_RES);

  // 1) activations -> bf16 ; weights -> bf16 fragment order
  k_f32_to_bf16<<<(kBN * kDIM) / 256, 256, 0, stream>>>(x, xb, kBN * kDIM);
  k_pack_w<<<(kDIM * 512) / 256, 256, 0, stream>>>(w_qs, wqsf, 512, kDIM * 512);
  k_pack_w<<<(kDIM * 512) / 256, 256, 0, stream>>>(w_ks, wksf, 512, kDIM * 512);
  k_pack_w<<<(kDIM * 512) / 256, 256, 0, stream>>>(w_vs, wvsf, 512, kDIM * 512);
  k_pack_w<<<(kDIM * 96) / 256, 256, 0, stream>>>(w_qp, wqpf, 96, kDIM * 96);
  k_pack_w<<<(kDIM * 96) / 256, 256, 0, stream>>>(w_kp, wkpf, 96, kDIM * 96);
  k_pack_w<<<(kDIM * 96) / 256, 256, 0, stream>>>(w_vp, wvpf, 96, kDIM * 96);
  k_pack_w<<<(kFIN * kDIM) / 256, 256, 0, stream>>>(w_o, wof, kDIM, kFIN * kDIM);

  // 2) projection GEMMs: 16x64 tiles (TN=4) for 512-wide, 16x32 (TN=2) for 96
  k_gemm<4><<<512, 128, 0, stream>>>(xb, wqsf, nullptr, nullptr, qaug, 1,
                                     kSCALAR_SCALE, kQK, kBN, 512, kDIM);
  k_gemm<4><<<512, 128, 0, stream>>>(xb, wksf, nullptr, nullptr, kaug, 1,
                                     1.0f, kQK, kBN, 512, kDIM);
  k_gemm<4><<<512, 128, 0, stream>>>(xb, wvsf, nullptr, nullptr, vaug, 1,
                                     1.0f, kVD, kBN, 512, kDIM);
  k_gemm<2><<<192, 128, 0, stream>>>(xb, wqpf, nullptr, tqp, nullptr, 0,
                                     1.0f, 0, kBN, 96, kDIM);
  k_gemm<2><<<192, 128, 0, stream>>>(xb, wkpf, nullptr, tkp, nullptr, 0,
                                     1.0f, 0, kBN, 96, kDIM);
  k_gemm<2><<<192, 128, 0, stream>>>(xb, wvpf, nullptr, tvp, nullptr, 0,
                                     1.0f, 0, kBN, 96, kDIM);

  // 3) rigid transform of points + augmented point slices + norm biases
  k_assemble_points<<<(kBN * kH) / 256, 256, 0, stream>>>(
      tqp, tkp, tvp, rot, trans, pwts, qaug, kaug, vaug, sqk, sqq);

  // 4) repack K/V into async-copy-friendly fragment slabs
  k_pack_kfrag<<<(kB * kH * kN * kQK) / 256, 256, 0, stream>>>(kaug, kfrag);
  k_pack_vfrag<<<(kB * kH * kN * kVD) / 256, 256, 0, stream>>>(vaug, vfrag);

  // 5) fused flash attention (2048 query tiles, 4 waves/block)
  k_attention<<<512, 128, 0, stream>>>(qaug, kfrag, vfrag, sqk, sqq, pwts, resb);

  // 6) inverse transform + norms + concat matrix
  k_finalize<<<(kBN * kH) / 256, 256, 0, stream>>>(resb, rot, trans, afin);

  // 7) output projection with bias
  k_gemm<4><<<512, 128, 0, stream>>>(afin, wof, b_o, out, nullptr, 0,
                                     1.0f, 0, kBN, kDIM, kFIN);
}
